// GraphGRUCell_38302518346045
// MI455X (gfx1250) — compile-verified, matrix operand order checked by reading, not run on hardware
//
#include <hip/hip_runtime.h>

#define B_  256
#define N_  240
#define E_  4338
#define F_  28
#define U_  100
#define K_  128            // U+F
#define EN  (E_ + N_)      // 4578 edges incl. self loops
#define NEG_SLOPE 0.2f

typedef __attribute__((ext_vector_type(16))) _Float16 v16h;
typedef __attribute__((ext_vector_type(8)))  float    v8f;

// f16 weight cache in workspace, laid out W[n][k] (k contiguous) so a WMMA
// B-fragment is two contiguous 16B loads per lane. N padded to tile multiple.
#define WG_OFF   0
#define WG_ROWS  112
#define WRU_OFF  (WG_ROWS * K_)
#define WRU_ROWS 208
#define WC_OFF   (WRU_OFF + WRU_ROWS * K_)
#define WC_ROWS  112
#define WTOT     (WC_OFF + WC_ROWS * K_)

// LDS partition (bytes), total 279,440 of the 320KB WGP LDS
#define SX_OFF    0              // 240x128 f16  (x -> cat1 -> cat2)
#define SH_OFF    61440          // 240x208 f16  (h -> r|u gates)
#define SAGG_OFF  161280         // 240x100 f32  (agg -> gstate = st)
#define SEDGE_OFF 257280         // 4578 f32 edge scratch
#define SASRC_OFF 275600         // 240 f32
#define SADST_OFF 276560         // 240 f32
#define SMAX_OFF  277520         // 240 f32
#define SDEN_OFF  278480         // 240 f32
#define SMEM_TOT  279440

__global__ void __launch_bounds__(256)
prep_weights(const float* __restrict__ Wg, const float* __restrict__ Wru,
             const float* __restrict__ Wc, _Float16* __restrict__ w) {
  for (int i = blockIdx.x * blockDim.x + threadIdx.x; i < WTOT;
       i += gridDim.x * blockDim.x) {
    if (i < (int)WRU_OFF) {                 // Wg is already [n][k] row-major
      int n = i / K_, k = i % K_;
      w[i] = (n < U_) ? (_Float16)Wg[n * K_ + k] : (_Float16)0.f;
    } else if (i < (int)WC_OFF) {           // W_ru is [k][2U]: transpose
      int j = i - WRU_OFF; int n = j / K_, k = j % K_;
      w[i] = (n < 2 * U_) ? (_Float16)Wru[k * (2 * U_) + n] : (_Float16)0.f;
    } else {                                // W_c is [k][U]: transpose
      int j = i - WC_OFF; int n = j / K_, k = j % K_;
      w[i] = (n < U_) ? (_Float16)Wc[k * U_ + n] : (_Float16)0.f;
    }
  }
}

// 16-bit A/B fragment per CDNA5 ISA striping: lanes 0-15 hold K {0..7,16..23},
// lanes 16-31 hold K {8..15,24..31} -> two contiguous 8xf16 (16B) chunks/lane.
__device__ __forceinline__ v16h ldfrag(const _Float16* rowptr, int half) {
  union { v16h v; uint4 q[2]; } u;
  u.q[0] = *(const uint4*)(rowptr + half * 8);
  u.q[1] = *(const uint4*)(rowptr + 16 + half * 8);
  return u.v;
}

// N-tile-stationary GEMM sweep: each wave owns N-tile columns, preloads the
// four K-step B fragments (32 VGPRs) from global once, then sweeps all 15
// M-tiles with only ds_load_b128 pairs feeding the v_wmma chain.
template <typename Epi>
__device__ __forceinline__ void gemm_nfixed(const _Float16* sA,
                                            const _Float16* __restrict__ wB,
                                            int ntiles, int lane, int wave,
                                            Epi epi) {
  const int half = lane >> 4;
  for (int nt = wave; nt < ntiles; nt += 8) {
    const _Float16* brow = wB + (nt * 16 + (lane & 15)) * K_;
    v16h b0 = ldfrag(brow + 0, half);
    v16h b1 = ldfrag(brow + 32, half);
    v16h b2 = ldfrag(brow + 64, half);
    v16h b3 = ldfrag(brow + 96, half);
    for (int mt = 0; mt < 15; ++mt) {
      const _Float16* arow = sA + (mt * 16 + (lane & 15)) * K_;
      v8f acc = {};
      acc = __builtin_amdgcn_wmma_f32_16x16x32_f16(false, ldfrag(arow + 0, half),
                                                   false, b0, (short)0, acc,
                                                   false, false);
      acc = __builtin_amdgcn_wmma_f32_16x16x32_f16(false, ldfrag(arow + 32, half),
                                                   false, b1, (short)0, acc,
                                                   false, false);
      acc = __builtin_amdgcn_wmma_f32_16x16x32_f16(false, ldfrag(arow + 64, half),
                                                   false, b2, (short)0, acc,
                                                   false, false);
      acc = __builtin_amdgcn_wmma_f32_16x16x32_f16(false, ldfrag(arow + 96, half),
                                                   false, b3, (short)0, acc,
                                                   false, false);
      epi(mt * 16 + 8 * half, nt * 16 + (lane & 15), acc);
    }
  }
}

__device__ __forceinline__ void atomicMaxF(float* addr, float val) {
  if (val >= 0.f) atomicMax((int*)addr, __float_as_int(val));
  else            atomicMin((unsigned int*)addr, __float_as_uint(val));
}

__global__ void __launch_bounds__(256)
gatgru_kernel(const float* __restrict__ inputs, const float* __restrict__ state,
              const int* __restrict__ esrc, const int* __restrict__ edst,
              const float* __restrict__ att_src, const float* __restrict__ att_dst,
              const float* __restrict__ gat_bias, const float* __restrict__ bias1,
              const float* __restrict__ b_ru, const float* __restrict__ b_c,
              const _Float16* __restrict__ w, float* __restrict__ out) {
  extern __shared__ char smem[];
  _Float16* sX   = (_Float16*)(smem + SX_OFF);
  _Float16* sH   = (_Float16*)(smem + SH_OFF);
  float*    sAgg = (float*)(smem + SAGG_OFF);
  float*    sEdg = (float*)(smem + SEDGE_OFF);
  float*    sAs  = (float*)(smem + SASRC_OFF);
  float*    sAd  = (float*)(smem + SADST_OFF);
  float*    sMx  = (float*)(smem + SMAX_OFF);
  float*    sDn  = (float*)(smem + SDEN_OFF);

  const int b    = blockIdx.x;
  const int tid  = threadIdx.x;
  const int lane = tid & 31, wave = tid >> 5;

  // L2 prefetch of the (shared, tiny) weight blocks -> global_prefetch_b8
  __builtin_prefetch(w + WG_OFF, 0, 3);
  __builtin_prefetch(w + WRU_OFF, 0, 3);
  __builtin_prefetch(w + WC_OFF, 0, 3);

  // ---- Phase 0: x = [state | inputs] -> f16 LDS; init reductions ----
  const float* st_b = state  + (size_t)b * N_ * U_;
  const float* xi_b = inputs + (size_t)b * N_ * F_;
  for (int i = tid; i < N_ * K_; i += 256) {
    int n = i / K_, f = i % K_;
    float v = (f < U_) ? st_b[n * U_ + f] : xi_b[n * F_ + (f - U_)];
    sX[i] = (_Float16)v;
  }
  for (int i = tid; i < N_; i += 256) { sMx[i] = -3.402823466e38f; sDn[i] = 0.f; }
  for (int i = tid; i < N_ * U_; i += 256) sAgg[i] = 0.f;
  __syncthreads();

  // ---- Phase 1: h = x @ Wg^T  (7 N-tiles x 15 M-tiles) ----
  gemm_nfixed(sX, w + WG_OFF, 7, lane, wave,
              [&](int m0, int n0, v8f acc) {
#pragma unroll
                for (int r = 0; r < 8; ++r)
                  sH[(m0 + r) * 208 + n0] = (_Float16)acc[r];
              });
  __syncthreads();

  // ---- Phase 2: attention logits per node ----
  for (int n = tid; n < N_; n += 256) {
    float as = 0.f, ad = 0.f;
    for (int u = 0; u < U_; ++u) {
      float hv = (float)sH[n * 208 + u];
      as += hv * att_src[u];
      ad += hv * att_dst[u];
    }
    sAs[n] = as; sAd[n] = ad;
  }
  __syncthreads();

  // ---- Phase 3: edge logits + segment max ----
  for (int e = tid; e < EN; e += 256) {
    int s = (e < E_) ? esrc[e] : (e - E_);
    int d = (e < E_) ? edst[e] : (e - E_);
    float v = sAs[s] + sAd[d];
    v = (v > 0.f) ? v : NEG_SLOPE * v;
    sEdg[e] = v;
    atomicMaxF(&sMx[d], v);
  }
  __syncthreads();

  // ---- Phase 4: exp + segment sum ----
  for (int e = tid; e < EN; e += 256) {
    int d = (e < E_) ? edst[e] : (e - E_);
    float ex = __expf(sEdg[e] - sMx[d]);
    sEdg[e] = ex;
    atomicAdd(&sDn[d], ex);
  }
  __syncthreads();

  // ---- Phase 5: message aggregation (alpha-weighted scatter-add) ----
  for (int idx = tid; idx < EN * 4; idx += 256) {
    int e = idx >> 2, part = idx & 3;
    int s = (e < E_) ? esrc[e] : (e - E_);
    int d = (e < E_) ? edst[e] : (e - E_);
    float alpha = sEdg[e] / sDn[d];
    int u0 = part * 25;
    for (int u = u0; u < u0 + 25; ++u)
      atomicAdd(&sAgg[d * U_ + u], alpha * (float)sH[s * 208 + u]);
  }
  __syncthreads();

  // ---- Phase 6: gstate = agg + biases; build cat1 = [xi | gstate] in sX ----
  for (int n = tid; n < N_; n += 256) {
    _Float16 xi[F_];
#pragma unroll
    for (int j = 0; j < F_; ++j) xi[j] = sX[n * K_ + U_ + j];
#pragma unroll
    for (int j = 0; j < F_; ++j) sX[n * K_ + j] = xi[j];
    for (int u = 0; u < U_; ++u) {
      float g = sAgg[n * U_ + u] + gat_bias[u] + bias1[u];
      sAgg[n * U_ + u] = g;                       // st, kept in f32
      sX[n * K_ + F_ + u] = (_Float16)g;
    }
  }
  __syncthreads();

  // ---- Phase 7: ru = sigmoid(cat1 @ W_ru + b_ru)  (13 N-tiles) ----
  gemm_nfixed(sX, w + WRU_OFF, 13, lane, wave,
              [&](int m0, int n0, v8f acc) {
                if (n0 < 2 * U_) {
                  float bb = b_ru[n0];
#pragma unroll
                  for (int r = 0; r < 8; ++r) {
                    float g = 1.f / (1.f + __expf(-(acc[r] + bb)));
                    sH[(m0 + r) * 208 + n0] = (_Float16)g; // 0..99=r,100..199=u
                  }
                }
              });
  __syncthreads();

  // ---- Phase 8: cat2 = [xi | r*st] in sX ----
  for (int n = tid; n < N_; n += 256)
    for (int u = 0; u < U_; ++u)
      sX[n * K_ + F_ + u] =
          (_Float16)((float)sH[n * 208 + u] * sAgg[n * U_ + u]);
  __syncthreads();

  // ---- Phase 9: c = tanh(cat2 @ W_c + b_c); out = u*st + (1-u)*c ----
  float* out_b = out + (size_t)b * N_ * U_;
  gemm_nfixed(sX, w + WC_OFF, 7, lane, wave,
              [&](int m0, int n0, v8f acc) {
                if (n0 < U_) {
                  float bb = b_c[n0];
#pragma unroll
                  for (int r = 0; r < 8; ++r) {
                    int m = m0 + r;
                    float c  = tanhf(acc[r] + bb);
                    float ug = (float)sH[m * 208 + U_ + n0];
                    float sv = sAgg[m * U_ + n0];
                    out_b[m * U_ + n0] = ug * sv + (1.f - ug) * c;
                  }
                }
              });
}

extern "C" void kernel_launch(void* const* d_in, const int* in_sizes, int n_in,
                              void* d_out, int out_size, void* d_ws, size_t ws_size,
                              hipStream_t stream) {
  (void)in_sizes; (void)n_in; (void)out_size; (void)ws_size;
  const float* inputs   = (const float*)d_in[0];
  const float* state    = (const float*)d_in[1];
  const int*   esrc     = (const int*)d_in[2];
  const int*   edst     = (const int*)d_in[3];
  const float* Wg       = (const float*)d_in[4];
  const float* att_src  = (const float*)d_in[5];
  const float* att_dst  = (const float*)d_in[6];
  const float* gat_bias = (const float*)d_in[7];
  const float* bias1    = (const float*)d_in[8];
  const float* W_ru     = (const float*)d_in[9];
  const float* b_ru     = (const float*)d_in[10];
  const float* W_c      = (const float*)d_in[11];
  const float* b_c      = (const float*)d_in[12];
  _Float16* w = (_Float16*)d_ws;   // needs 110,592 B of workspace

  prep_weights<<<64, 256, 0, stream>>>(Wg, W_ru, W_c, w);
  gatgru_kernel<<<B_, 256, SMEM_TOT, stream>>>(
      inputs, state, esrc, edst, att_src, att_dst, gat_bias, bias1,
      b_ru, b_c, w, (float*)d_out);
}
